// MultiViewContrastiveModel_50611894616716
// MI455X (gfx1250) — matrix operand order repaired and checked by reading, not compiled, not run on hardware
//
#include <hip/hip_runtime.h>
#include <hip/hip_bf16.h>

// ---------------------------------------------------------------------------
// MultiViewContrastiveModel on MI455X (gfx1250).
// Dense GEMMs: v_wmma_f32_16x16x32_bf16, 4 WMMAs per wave per k-step
// (16x64 output strip per wave, 128x64 per block). fp32 inputs converted to
// bf16 on the fly, fp32 accumulation. SpMM is a flat atomic scatter.
// ---------------------------------------------------------------------------

typedef __attribute__((ext_vector_type(16))) __bf16 v16bf;
typedef __attribute__((ext_vector_type(8)))  float  v8f;

#define LRELU(x) ((x) > 0.0f ? (x) : 0.2f * (x))

static __device__ __forceinline__ const float* addrA(const float* __restrict__ A, int m, int k,
                                                     int lda, int a_kblock, size_t a_vstride) {
  if (a_kblock) {
    // "concat over views" addressing: A[m, k] = z[(k>>7)][m][k&127]
    return A + (size_t)(k >> 7) * a_vstride + (size_t)m * lda + (k & 127);
  }
  return A + (size_t)m * lda + k;
}

// C[M,Nc] = act(A[M,K] @ B[K,Nc] + bias[Nc])
// Block = 256 threads = 8 waves. Block tile 128(M) x 64(N); each wave owns a
// 16x64 strip = 4 WMMA accumulators. B k-slab (32x64) staged in LDS as bf16,
// column-major so a B fragment is one contiguous 32B chunk per lane.
__global__ __launch_bounds__(256)
void gemm_bias_act(const float* __restrict__ A, size_t a_vstride, int a_kblock, int lda,
                   const float* __restrict__ B, int ldb,
                   const float* __restrict__ bias,
                   float* __restrict__ C, int ldc,
                   int M, int K, int act) {
  __shared__ __bf16 sB[64 * 32];   // sB[col*32 + kk] holds k = kk with bits 3,4 swapped; col in [0,64)
  const int lane = threadIdx.x & 31;
  const int wave = threadIdx.x >> 5;
  const int half = lane >> 4;      // lanes 16..31 hold K 8..15 and 24..31
  const int ml   = lane & 15;
  const int row0 = blockIdx.x * 128 + wave * 16;
  const int col0 = blockIdx.y * 64;
  int m = row0 + ml;
  if (m >= M) m = M - 1;           // clamp loads; stores guarded (A row m only feeds D row m)

  v8f acc0 = {}, acc1 = {}, acc2 = {}, acc3 = {};
  for (int k0 = 0; k0 < K; k0 += 32) {
    // cooperative stage of B[k0:k0+32, col0:col0+64] -> LDS (bf16, column-major)
    for (int t = threadIdx.x; t < 2048; t += 256) {
      int nn = t >> 5, kk = t & 31;
      sB[t] = (__bf16)B[(size_t)(k0 + ((kk & 7) | ((kk & 8) << 1) | ((kk & 16) >> 1))) * ldb + col0 + nn];
    }
    __syncthreads();

    // A fragment, 16-bit 16x32 layout:
    // lanes 0-15: K k0+0..7 / k0+16..23;  lanes 16-31: K k0+8..15 / k0+24..31
    const float* ap0 = addrA(A, m, k0 + half * 8,      lda, a_kblock, a_vstride);
    const float* ap1 = addrA(A, m, k0 + 16 + half * 8, lda, a_kblock, a_vstride);
    float4 a0 = ((const float4*)ap0)[0];
    float4 a1 = ((const float4*)ap0)[1];
    float4 a2 = ((const float4*)ap1)[0];
    float4 a3 = ((const float4*)ap1)[1];
    v16bf a;
    a[0]  = (__bf16)a0.x; a[1]  = (__bf16)a0.y; a[2]  = (__bf16)a0.z; a[3]  = (__bf16)a0.w;
    a[4]  = (__bf16)a1.x; a[5]  = (__bf16)a1.y; a[6]  = (__bf16)a1.z; a[7]  = (__bf16)a1.w;
    a[8]  = (__bf16)a2.x; a[9]  = (__bf16)a2.y; a[10] = (__bf16)a2.z; a[11] = (__bf16)a2.w;
    a[12] = (__bf16)a3.x; a[13] = (__bf16)a3.y; a[14] = (__bf16)a3.z; a[15] = (__bf16)a3.w;

    if (k0 + 32 < K) {  // prefetch next A k-slab (global_prefetch_b8)
      __builtin_prefetch(addrA(A, m, k0 + 32 + half * 8, lda, a_kblock, a_vstride), 0, 3);
    }

    // B fragments: lane (half, ml) of N-subtile nt reads K = half*8..+7 and 16+half*8..+7 of
    // column nt*16+ml -> contiguous 16 bf16 (32B, aligned) in LDS.
    v16bf b0 = *(const v16bf*)&sB[(0 * 16 + ml) * 32 + half * 16];
    v16bf b1 = *(const v16bf*)&sB[(1 * 16 + ml) * 32 + half * 16];
    v16bf b2 = *(const v16bf*)&sB[(2 * 16 + ml) * 32 + half * 16];
    v16bf b3 = *(const v16bf*)&sB[(3 * 16 + ml) * 32 + half * 16];

    acc0 = __builtin_amdgcn_wmma_f32_16x16x32_bf16(false, a, false, b0, (short)0, acc0, false, false);
    acc1 = __builtin_amdgcn_wmma_f32_16x16x32_bf16(false, a, false, b1, (short)0, acc1, false, false);
    acc2 = __builtin_amdgcn_wmma_f32_16x16x32_bf16(false, a, false, b2, (short)0, acc2, false, false);
    acc3 = __builtin_amdgcn_wmma_f32_16x16x32_bf16(false, a, false, b3, (short)0, acc3, false, false);
    __syncthreads();
  }

  // D layout: lane l holds column (l&15) of its subtile; VGPR r -> row r (+8 for lanes 16-31)
#pragma unroll
  for (int nt = 0; nt < 4; ++nt) {
    const v8f accv = nt == 0 ? acc0 : nt == 1 ? acc1 : nt == 2 ? acc2 : acc3;
    const int n = col0 + nt * 16 + ml;
    const float bv = bias ? bias[n] : 0.0f;
#pragma unroll
    for (int r = 0; r < 8; ++r) {
      int mo = row0 + half * 8 + r;
      if (mo < M) {
        float v = accv[r] + bv;
        if (act == 1)      v = fmaxf(v, 0.0f);
        else if (act == 2) v = LRELU(v);
        C[(size_t)mo * ldc + n] = v;
      }
    }
  }
}

// --------------------------- graph kernels ---------------------------------

__global__ void zero_f(float* __restrict__ p, unsigned n) {
  unsigned i = blockIdx.x * blockDim.x + threadIdx.x;
  if (i < n) p[i] = 0.0f;
}

__global__ void degree_accum(const int* __restrict__ rows, const float* __restrict__ vals,
                             float* __restrict__ deg, int E, int N) {
  int e = blockIdx.x * blockDim.x + threadIdx.x;
  if (e >= E + N) return;
  if (e < E) atomicAdd(&deg[rows[e]], vals[e]);
  else       atomicAdd(&deg[e - E], 1.0f);   // self loop
}

__global__ void deg_to_dinv(float* __restrict__ deg, int N) {
  int i = blockIdx.x * blockDim.x + threadIdx.x;
  if (i < N) deg[i] = rsqrtf(fmaxf(deg[i], 1e-12f));
}

// y[row[e], :] += val[e]*dinv[row]*dinv[col] * x[col[e], :], F = 1<<logF
__global__ void spmm_scatter(const int* __restrict__ rows, const int* __restrict__ cols,
                             const float* __restrict__ vals, const float* __restrict__ dinv,
                             const float* __restrict__ x, float* __restrict__ y,
                             int E, int N, int logF) {
  const int F = 1 << logF;
  unsigned idx = blockIdx.x * blockDim.x + threadIdx.x;
  unsigned total = (unsigned)(E + N) << logF;
  if (idx >= total) return;
  int e = idx >> logF;
  int f = idx & (F - 1);
  int r, c; float w;
  if (e < E) { r = rows[e]; c = cols[e]; w = vals[e]; }
  else       { r = e - E;   c = r;       w = 1.0f;    }
  w *= dinv[r] * dinv[c];
  atomicAdd(&y[((size_t)r << logF) | f], w * x[((size_t)c << logF) | f]);
}

__global__ void leaky_inplace(float* __restrict__ p, unsigned n) {
  unsigned i = blockIdx.x * blockDim.x + threadIdx.x;
  if (i < n) p[i] = LRELU(p[i]);
}

// out[n] = b[0] + dot(t[n, 0:64], w[0:64])
__global__ void head_logit(const float* __restrict__ t, const float* __restrict__ w,
                           const float* __restrict__ b, float* __restrict__ out, int N) {
  int n_ = blockIdx.x * blockDim.x + threadIdx.x;
  if (n_ >= N) return;
  float s = b[0];
  const float* tp = t + (size_t)n_ * 64;
#pragma unroll
  for (int j = 0; j < 64; ++j) s += tp[j] * w[j];
  out[n_] = s;
}

// att[n, 0:3] = softmax(hidden[n, 0:128] @ W2[128,3] + b2)
__global__ void att_softmax(const float* __restrict__ hidden, const float* __restrict__ W2,
                            const float* __restrict__ b2, float* __restrict__ att, int N) {
  int n_ = blockIdx.x * blockDim.x + threadIdx.x;
  if (n_ >= N) return;
  float s0 = b2[0], s1 = b2[1], s2 = b2[2];
  const float* h = hidden + (size_t)n_ * 128;
#pragma unroll 4
  for (int k = 0; k < 128; ++k) {
    float hv = h[k];
    s0 += hv * W2[k * 3 + 0];
    s1 += hv * W2[k * 3 + 1];
    s2 += hv * W2[k * 3 + 2];
  }
  float mx = fmaxf(s0, fmaxf(s1, s2));
  float e0 = __expf(s0 - mx), e1 = __expf(s1 - mx), e2 = __expf(s2 - mx);
  float inv = 1.0f / (e0 + e1 + e2);
  att[(size_t)n_ * 3 + 0] = e0 * inv;
  att[(size_t)n_ * 3 + 1] = e1 * inv;
  att[(size_t)n_ * 3 + 2] = e2 * inv;
}

// fused[n, d] = sum_v att[n, v] * z[v, n, d]
__global__ void fuse_z(const float* __restrict__ z, const float* __restrict__ att,
                       float* __restrict__ fused, int N) {
  unsigned i = blockIdx.x * blockDim.x + threadIdx.x;
  if (i >= (unsigned)N * 128u) return;
  int n_ = i >> 7, d = i & 127;
  size_t nd = (size_t)N * 128;
  size_t off = (size_t)n_ * 128 + d;
  fused[i] = att[(size_t)n_ * 3 + 0] * z[off]
           + att[(size_t)n_ * 3 + 1] * z[nd + off]
           + att[(size_t)n_ * 3 + 2] * z[2 * nd + off];
}

// one wave (32 lanes) per row of 128 floats; wave32 shuffle reduction
__global__ __launch_bounds__(256)
void l2_normalize_rows(float* __restrict__ p, int N) {
  int wave = threadIdx.x >> 5, lane = threadIdx.x & 31;
  int n_ = blockIdx.x * 8 + wave;
  if (n_ >= N) return;
  float4* rp = (float4*)(p + (size_t)n_ * 128);
  float4 v = rp[lane];
  float s = v.x * v.x + v.y * v.y + v.z * v.z + v.w * v.w;
#pragma unroll
  for (int off = 16; off; off >>= 1) s += __shfl_xor(s, off, 32);
  float inv = 1.0f / fmaxf(sqrtf(s), 1e-12f);
  v.x *= inv; v.y *= inv; v.z *= inv; v.w *= inv;
  rp[lane] = v;
}

// ---------------------------------------------------------------------------

extern "C" void kernel_launch(void* const* d_in, const int* in_sizes, int n_in,
                              void* d_out, int out_size, void* d_ws, size_t ws_size,
                              hipStream_t stream) {
  (void)in_sizes; (void)n_in; (void)out_size; (void)ws_size;
  const int N = 50000, V = 3, E = 800000;
  const int IN = 256, H1 = 256, H2 = 128, D = 128, CH = 64;

  const float* X      = (const float*)d_in[0];
  const int*   rows   = (const int*)  d_in[1];
  const int*   cols   = (const int*)  d_in[2];
  const float* scores = (const float*)d_in[3];
  const float* gc1W = (const float*)d_in[4],  *gc1b = (const float*)d_in[5];
  const float* gc2W = (const float*)d_in[6],  *gc2b = (const float*)d_in[7];
  const float* gc3W = (const float*)d_in[8],  *gc3b = (const float*)d_in[9];
  const float* cW1  = (const float*)d_in[10], *cb1  = (const float*)d_in[11];
  const float* cW2  = (const float*)d_in[12], *cb2  = (const float*)d_in[13];
  const float* pW1  = (const float*)d_in[14], *pb1  = (const float*)d_in[15];
  const float* pW2  = (const float*)d_in[16], *pb2  = (const float*)d_in[17];
  const float* aW1  = (const float*)d_in[18], *ab1  = (const float*)d_in[19];
  const float* aW2  = (const float*)d_in[20], *ab2  = (const float*)d_in[21];
  const float* fW1  = (const float*)d_in[22], *fb1  = (const float*)d_in[23];
  const float* fW2  = (const float*)d_in[24], *fb2  = (const float*)d_in[25];

  // d_out regions (return order): fused_logit, fused, att, z, view_logits, proj
  float* out      = (float*)d_out;
  float* o_flogit = out;
  float* o_fused  = out + (size_t)N;
  float* o_att    = out + (size_t)N * 129;
  float* o_z      = out + (size_t)N * 132;
  float* o_vlogit = out + (size_t)N * 516;
  float* o_proj   = out + (size_t)N * 519;

  // workspace: deg/dinv (in place) + two ping-pong N x 256 buffers
  float* deg  = (float*)d_ws;
  float* bufA = deg + N;
  float* bufB = bufA + (size_t)N * 256;

  auto gemm = [&](const float* A, size_t avs, int akb, int lda,
                  const float* B, int Nc, const float* bias,
                  float* C, int K, int act) {
    dim3 grid((N + 127) / 128, Nc / 64);
    gemm_bias_act<<<grid, 256, 0, stream>>>(A, avs, akb, lda, B, Nc, bias, C, Nc, N, K, act);
  };
  auto zero = [&](float* p, unsigned n) {
    zero_f<<<(n + 255) / 256, 256, 0, stream>>>(p, n);
  };
  auto leaky = [&](float* p, unsigned n) {
    leaky_inplace<<<(n + 255) / 256, 256, 0, stream>>>(p, n);
  };

  const unsigned EN = (unsigned)(E + N);

  for (int v = 0; v < V; ++v) {
    const int*   rv = rows   + (size_t)v * E;
    const int*   cv = cols   + (size_t)v * E;
    const float* sv = scores + (size_t)v * E;

    // symmetric normalization: deg -> d^-1/2 (in place)
    zero(deg, (unsigned)N);
    degree_accum<<<(EN + 255) / 256, 256, 0, stream>>>(rv, sv, deg, E, N);
    deg_to_dinv<<<(N + 255) / 256, 256, 0, stream>>>(deg, N);

    // layer 1: leaky(spmm(X @ W1 + b1)), F = 256
    gemm(X, 0, 0, IN, gc1W + (size_t)v * IN * H1, H1, gc1b + (size_t)v * H1, bufA, IN, 0);
    zero(bufB, (unsigned)N * H1);
    spmm_scatter<<<((EN << 8) + 255) / 256, 256, 0, stream>>>(rv, cv, sv, deg, bufA, bufB, E, N, 8);
    leaky(bufB, (unsigned)N * H1);

    // layer 2: leaky(spmm(h1 @ W2 + b2)), F = 128
    gemm(bufB, 0, 0, H1, gc2W + (size_t)v * H1 * H2, H2, gc2b + (size_t)v * H2, bufA, H1, 0);
    zero(bufB, (unsigned)N * H2);
    spmm_scatter<<<((EN << 7) + 255) / 256, 256, 0, stream>>>(rv, cv, sv, deg, bufA, bufB, E, N, 7);
    leaky(bufB, (unsigned)N * H2);

    // layer 3: z = spmm(h2 @ W3 + b3) -> d_out z region
    gemm(bufB, 0, 0, H2, gc3W + (size_t)v * H2 * D, D, gc3b + (size_t)v * D, bufA, H2, 0);
    float* zv = o_z + (size_t)v * N * D;
    zero(zv, (unsigned)N * D);
    spmm_scatter<<<((EN << 7) + 255) / 256, 256, 0, stream>>>(rv, cv, sv, deg, bufA, zv, E, N, 7);

    // classifier head: relu(z @ cW1 + cb1) @ cW2 + cb2
    gemm(zv, 0, 0, D, cW1 + (size_t)v * D * CH, CH, cb1 + (size_t)v * CH, bufA, D, 1);
    head_logit<<<(N + 255) / 256, 256, 0, stream>>>(bufA, cW2 + (size_t)v * CH, cb2 + v,
                                                    o_vlogit + (size_t)v * N, N);

    // projection head: normalize(relu(z @ pW1 + pb1) @ pW2 + pb2)
    gemm(zv, 0, 0, D, pW1 + (size_t)v * D * D, D, pb1 + (size_t)v * D, bufA, D, 1);
    float* pv = o_proj + (size_t)v * N * D;
    gemm(bufA, 0, 0, D, pW2 + (size_t)v * D * D, D, pb2 + (size_t)v * D, pv, D, 0);
    l2_normalize_rows<<<(N + 7) / 8, 256, 0, stream>>>(pv, N);
  }

  // attention fusion: hidden = relu(concat(z) @ aW1 + ab1), K = V*D = 384
  // (concat addressing reads z[V,N,128] directly, no transpose copy)
  gemm(o_z, (size_t)N * D, 128, D, aW1, D, ab1, bufA, V * D, 1);
  att_softmax<<<(N + 255) / 256, 256, 0, stream>>>(bufA, aW2, ab2, o_att, N);
  fuse_z<<<((unsigned)N * 128 + 255) / 256, 256, 0, stream>>>(o_z, o_att, o_fused, N);

  // fused classifier
  gemm(o_fused, 0, 0, D, fW1, CH, fb1, bufA, D, 1);
  head_logit<<<(N + 255) / 256, 256, 0, stream>>>(bufA, fW2, fb2, o_flogit, N);
}
